// Model_49031346651152
// MI455X (gfx1250) — matrix operand order repaired and checked by reading, not compiled, not run on hardware
//
#include <hip/hip_runtime.h>
#include <math.h>

// ---------------- model constants ----------------
#define BB   128          // batch
#define CC   64           // channels
#define LL   768          // seq len
#define PP   96           // patch len
#define NOUT 96           // head out
#define NP   8            // patches per channel (L/P)
#define DD   512          // model dim
#define HH   8            // heads
#define DHD  64           // head dim
#define DFF_ 2048
#define ELN  4            // layers
#define MH   32           // hyperedges
#define BCN  (BB*CC)      // 8192
#define TT   (BCN*NP)     // 65536 tokens

typedef _Float16 v16h __attribute__((ext_vector_type(16)));
typedef _Float16 v8h  __attribute__((ext_vector_type(8)));
typedef float    v8f  __attribute__((ext_vector_type(8)));

#if defined(__has_builtin)
#if __has_builtin(__builtin_amdgcn_sched_barrier)
#define K455_SCHED_FENCE() __builtin_amdgcn_sched_barrier(0)
#endif
#endif
#ifndef K455_SCHED_FENCE
#define K455_SCHED_FENCE() do {} while (0)
#endif

// ================= generic WMMA GEMM-NT =================
// C[M,N] = A[M,K](f16) * B[N,K](f16)^T  [+bias[N]] [+res[M,N](f32)] [gelu]
// writes fp32 (outF) and/or f16 (outH). K must be a multiple of 32.
// Block: 256 threads (8 waves), tile 64(M) x 128(N), K staged 32 at a time,
// double-buffered LDS, one barrier per K-chunk, vector (b128) staging copies.
// A sched fence between fragment loads and WMMAs forces all ds_loads to issue
// first (distinct registers per fragment), so the 4 WMMAs run back-to-back.
#define GBM 64
#define GBN 128
#define GBK 32
#define LDK (GBK + 8)
__global__ __launch_bounds__(256) void k455_gemm_nt(
    const _Float16* __restrict__ A, const _Float16* __restrict__ B,
    float* __restrict__ outF, _Float16* __restrict__ outH,
    const float* __restrict__ bias, const float* __restrict__ res,
    int M, int N, int K, int actGelu)
{
    __shared__ _Float16 As[2][GBM][LDK];
    __shared__ _Float16 Bs[2][GBN][LDK];
    const int tid  = threadIdx.x;
    const int lane = tid & 31;
    const int w    = tid >> 5;
    const int wm   = (w & 3) * 16;   // wave M offset in tile
    const int wn   = (w >> 2) * 64;  // wave N offset in tile (4 x 16 sub-tiles)
    const int bm   = blockIdx.y * GBM;
    const int bn   = blockIdx.x * GBN;
    const int hi   = lane >> 4;
    const int lo   = lane & 15;

    // staging assignments (vectorized 16B copies)
    const int arow = tid >> 2, acol = (tid & 3) * 8;   // A: 64 rows x 32 cols
    const int brow = tid >> 1, bcol = (tid & 1) * 16;  // B: 128 rows x 32 cols

    v8f acc[4];
    #pragma unroll
    for (int t = 0; t < 4; ++t) acc[t] = (v8f){};

    // ---- prologue: load chunk 0 into regs, stage to LDS buf 0 ----
    v8h ra = {};
    v8h rb0 = {}, rb1 = {};
    {
        int gr = bm + arow;
        if (gr < M) ra = *(const v8h*)(A + (size_t)gr * K + acol);
        int hr = bn + brow;
        if (hr < N) {
            const _Float16* src = B + (size_t)hr * K + bcol;
            rb0 = *(const v8h*)src;
            rb1 = *(const v8h*)(src + 8);
        }
    }
    *(v8h*)&As[0][arow][acol]     = ra;
    *(v8h*)&Bs[0][brow][bcol]     = rb0;
    *(v8h*)&Bs[0][brow][bcol + 8] = rb1;
    __syncthreads();

    int buf = 0;
    for (int k0 = 0; k0 < K; k0 += GBK) {
        const bool more = (k0 + GBK) < K;
        // ---- issue global loads for next chunk (latency hides under WMMA) ----
        if (more) {
            ra = (v8h){}; rb0 = (v8h){}; rb1 = (v8h){};
            int gr = bm + arow;
            if (gr < M) ra = *(const v8h*)(A + (size_t)gr * K + k0 + GBK + acol);
            int hr = bn + brow;
            if (hr < N) {
                const _Float16* src = B + (size_t)hr * K + k0 + GBK + bcol;
                rb0 = *(const v8h*)src;
                rb1 = *(const v8h*)(src + 8);
            }
        }
        // ---- all fragment loads first ----
        v16h a;
        #pragma unroll
        for (int e = 0; e < 16; ++e) {
            int ka = (e >> 3) * 16 + hi * 8 + ((e >> 1) & 3) * 2 + (e & 1);
            a[e] = As[buf][wm + lo][ka];
        }
        v16h bfr[4];
        #pragma unroll
        for (int t = 0; t < 4; ++t) {
            #pragma unroll
            for (int e = 0; e < 16; ++e)
                bfr[t][e] = Bs[buf][wn + t * 16 + lo][(e >> 3) * 16 + hi * 8 + (e & 7)];
        }
        // ---- fence: no WMMA may be hoisted between the ds_loads ----
        K455_SCHED_FENCE();
        // ---- 4 back-to-back WMMAs ----
        #pragma unroll
        for (int t = 0; t < 4; ++t)
            acc[t] = __builtin_amdgcn_wmma_f32_16x16x32_f16(false, a, false, bfr[t], (short)0, acc[t], false, false);
        // ---- stage next chunk into the other buffer ----
        if (more) {
            *(v8h*)&As[buf ^ 1][arow][acol]     = ra;
            *(v8h*)&Bs[buf ^ 1][brow][bcol]     = rb0;
            *(v8h*)&Bs[buf ^ 1][brow][bcol + 8] = rb1;
            __syncthreads();
        }
        buf ^= 1;
    }

    // ---- epilogue: C/D layout m = r + 8*hi, n = lo ----
    #pragma unroll
    for (int r = 0; r < 8; ++r) {
        int gm = bm + wm + r + 8 * hi;
        if (gm >= M) continue;
        #pragma unroll
        for (int t = 0; t < 4; ++t) {
            int gn = bn + wn + t * 16 + lo;
            if (gn >= N) continue;
            float vv = acc[t][r];
            if (bias) vv += bias[gn];
            if (res)  vv += res[(size_t)gm * N + gn];
            if (actGelu) {
                float x = vv;
                vv = 0.5f * x * (1.0f + tanhf(0.7978845608028654f * (x + 0.044715f * x * x * x)));
            }
            size_t idx = (size_t)gm * N + gn;
            if (outF) outF[idx] = vv;
            if (outH) outH[idx] = (_Float16)vv;
        }
    }
}

// ================= instance-norm stats over L per (b,c) =================
__global__ __launch_bounds__(256) void k455_stats(const float* __restrict__ x_enc,
                                                  float* __restrict__ means,
                                                  float* __restrict__ stdev)
{
    int b = blockIdx.x, tid = threadIdx.x;
    int c = tid & 63, qt = tid >> 6;           // 4 quarters of 192
    __shared__ float s1[256], s2[256];
    float s = 0.f, sq = 0.f;
    for (int l = qt * 192; l < qt * 192 + 192; ++l) {
        float v = x_enc[((size_t)b * LL + l) * CC + c];
        s += v; sq += v * v;
    }
    s1[tid] = s; s2[tid] = sq;
    __syncthreads();
    if (qt == 0) {
        float S  = s1[c] + s1[64 + c] + s1[128 + c] + s1[192 + c];
        float SQ = s2[c] + s2[64 + c] + s2[128 + c] + s2[192 + c];
        float mean = S / (float)LL;
        float var  = SQ / (float)LL - mean * mean;
        means[b * CC + c]  = mean;
        stdev[b * CC + c]  = sqrtf(var + 1e-5f);
    }
}

// ================= patchify (normalize + transpose) -> f16 =================
__global__ void k455_patchify(const float* __restrict__ x_enc, const float* __restrict__ means,
                              const float* __restrict__ stdev, _Float16* __restrict__ Xp, int total)
{
    int idx = blockIdx.x * blockDim.x + threadIdx.x;
    if (idx >= total) return;
    int t = idx / PP, p = idx % PP;
    int bc = t >> 3, n = t & 7;
    int b = bc >> 6, c = bc & 63;
    float v = (x_enc[((size_t)b * LL + n * PP + p) * CC + c] - means[bc]) / stdev[bc];
    Xp[idx] = (_Float16)v;
}

// ================= positional embedding add + f16 copy =================
__global__ void k455_add_pos(float* __restrict__ x, _Float16* __restrict__ xh, int total)
{
    int idx = blockIdx.x * blockDim.x + threadIdx.x;
    if (idx >= total) return;
    int t = idx >> 9, d = idx & 511;
    int n = t & 7;
    int i2 = (d >> 1) * 2;
    float div = expf(-(float)i2 * (9.210340371976184f / (float)DD)); // ln(10000)/D
    float ang = (float)n * div;
    float pe = (d & 1) ? cosf(ang) : sinf(ang);
    float v = x[idx] + pe;
    x[idx] = v; xh[idx] = (_Float16)v;
}

// ================= f32 -> f16 convert / transpose-convert =================
__global__ void k455_cvt(const float* __restrict__ s, _Float16* __restrict__ d, int n)
{
    int i = blockIdx.x * blockDim.x + threadIdx.x;
    if (i < n) d[i] = (_Float16)s[i];
}
__global__ void k455_cvt_t(const float* __restrict__ s, _Float16* __restrict__ d, int rows, int cols)
{
    int i = blockIdx.x * blockDim.x + threadIdx.x;
    if (i >= rows * cols) return;
    int r = i / cols, c = i % cols;
    d[(size_t)c * rows + r] = (_Float16)s[i];  // d[c][r] = s[r][c]
}

// ================= causal attention, T=8, per (bc,h) =================
__global__ __launch_bounds__(64) void k455_attn(const float* __restrict__ q, const float* __restrict__ k,
                                                const float* __restrict__ v, float* __restrict__ o,
                                                _Float16* __restrict__ oh)
{
    int bc = blockIdx.x, h = blockIdx.y, tid = threadIdx.x;
    __shared__ float qs[8][64], ks[8][64], vs[8][64], sS[8][8], aS[8][8];
    int i = tid >> 3, c0 = (tid & 7) * 8;
    size_t base = ((size_t)bc * NP) * DD + h * DHD;
    #pragma unroll
    for (int j = 0; j < 8; ++j) {
        size_t g = base + (size_t)i * DD + c0 + j;
        qs[i][c0 + j] = q[g]; ks[i][c0 + j] = k[g]; vs[i][c0 + j] = v[g];
    }
    __syncthreads();
    int qi = tid >> 3, kj = tid & 7;
    float s;
    if (kj > qi) s = -1e9f;
    else {
        s = 0.f;
        #pragma unroll
        for (int d = 0; d < 64; ++d) s += qs[qi][d] * ks[kj][d];
        s *= 0.125f;  // 1/sqrt(64)
    }
    sS[qi][kj] = s;
    __syncthreads();
    float mx = -1e30f;
    #pragma unroll
    for (int j = 0; j < 8; ++j) mx = fmaxf(mx, sS[qi][j]);
    float sum = 0.f;
    #pragma unroll
    for (int j = 0; j < 8; ++j) sum += expf(sS[qi][j] - mx);
    aS[qi][kj] = expf(sS[qi][kj] - mx) / sum;
    __syncthreads();
    #pragma unroll
    for (int dd = 0; dd < 8; ++dd) {
        float acc = 0.f;
        #pragma unroll
        for (int j = 0; j < 8; ++j) acc += aS[i][j] * vs[j][c0 + dd];
        size_t idx = base + (size_t)i * DD + c0 + dd;
        o[idx] = acc; oh[idx] = (_Float16)acc;
    }
}

// ================= row LayerNorm (D=512) =================
__global__ __launch_bounds__(128) void k455_ln(const float* __restrict__ in, const float* __restrict__ g,
                                               const float* __restrict__ b, float* __restrict__ outF,
                                               _Float16* __restrict__ outH, int D)
{
    int row = blockIdx.x, tid = threadIdx.x;
    __shared__ float red[128];
    const float* x = in + (size_t)row * D;
    float s = 0.f, sq = 0.f;
    for (int j = tid; j < D; j += 128) { float v = x[j]; s += v; sq += v * v; }
    red[tid] = s; __syncthreads();
    for (int st = 64; st > 0; st >>= 1) { if (tid < st) red[tid] += red[tid + st]; __syncthreads(); }
    float mean = red[0] / (float)D; __syncthreads();
    red[tid] = sq; __syncthreads();
    for (int st = 64; st > 0; st >>= 1) { if (tid < st) red[tid] += red[tid + st]; __syncthreads(); }
    float var = red[0] / (float)D - mean * mean;
    float rstd = rsqrtf(var + 1e-5f);
    for (int j = tid; j < D; j += 128) {
        float v = (x[j] - mean) * rstd * g[j] + b[j];
        size_t idx = (size_t)row * D + j;
        if (outF) outF[idx] = v;
        if (outH) outH[idx] = (_Float16)v;
    }
}

// ================= token-order -> node-order gather =================
__global__ void k455_gather(const float* __restrict__ src, float* __restrict__ dstF,
                            _Float16* __restrict__ dstH, int total)
{
    int idx = blockIdx.x * blockDim.x + threadIdx.x;
    if (idx >= total) return;
    int r = idx >> 9, d = idx & 511;     // node row, feature
    int g = r >> 7, bb = r & 127;        // graph, node(batch)
    int c = g >> 3, n = g & 7;
    int t = (bb * CC + c) * NP + n;
    float v = src[(size_t)t * DD + d];
    dstF[idx] = v; dstH[idx] = (_Float16)v;
}

// ================= node-order -> token-order scatter (f16) =================
__global__ void k455_scatter(const float* __restrict__ Yall, _Float16* __restrict__ ehH, int total)
{
    int idx = blockIdx.x * blockDim.x + threadIdx.x;
    if (idx >= total) return;
    int t = idx >> 9, d = idx & 511;
    int b = t >> 9;                       // t = (b*64+c)*8+n
    int c = (t >> 3) & 63, n = t & 7;
    int g = c * NP + n;
    int r = g * BB + b;
    ehH[idx] = (_Float16)Yall[(size_t)r * DD + d];
}

// ================= HGAT: scores[g,h,b,m] =================
__global__ __launch_bounds__(256) void k455_hgat_scores(const float* __restrict__ qAll,
                                                        const float* __restrict__ kE,
                                                        float* __restrict__ scores)
{
    int g = blockIdx.x, h = blockIdx.y, tid = threadIdx.x;
    __shared__ float ks[MH][DHD];
    for (int p = tid; p < MH * DHD; p += 256) {
        int m = p >> 6, d = p & 63;
        ks[m][d] = kE[(size_t)m * DD + h * DHD + d];
    }
    __syncthreads();
    for (int p = tid; p < 128 * MH; p += 256) {
        int b = p >> 5, m = p & 31;
        const float* qp = qAll + ((size_t)g * BB + b) * DD + h * DHD;
        float acc = 0.f;
        #pragma unroll
        for (int d = 0; d < 64; ++d) acc += qp[d] * ks[m][d];
        scores[(((size_t)g * HH + h) * BB + b) * MH + m] = acc * 0.125f;
    }
}

// ================= HGAT: sm = mean_h scores; thr = median over nodes =================
__global__ __launch_bounds__(128) void k455_hgat_smthr(const float* __restrict__ scores,
                                                       float* __restrict__ sm, float* __restrict__ thr)
{
    int g = blockIdx.x, b = threadIdx.x;
    __shared__ float smL[128][MH + 1];
    __shared__ float c63[MH], c64[MH];
    for (int m = 0; m < MH; ++m) {
        float acc = 0.f;
        #pragma unroll
        for (int h = 0; h < HH; ++h) acc += scores[(((size_t)g * HH + h) * BB + b) * MH + m];
        float v = acc * (1.0f / HH);
        smL[b][m] = v;
        sm[((size_t)g * BB + b) * MH + m] = v;
    }
    __syncthreads();
    for (int m = 0; m < MH; ++m) {
        float val = smL[b][m];
        int less = 0, eq = 0;
        for (int j = 0; j < 128; ++j) {
            float x = smL[j][m];
            less += (x < val); eq += (x == val);
        }
        if (less <= 63 && 63 < less + eq) c63[m] = val;  // order stat 63
        if (less <= 64 && 64 < less + eq) c64[m] = val;  // order stat 64
    }
    __syncthreads();
    if (b < MH) thr[(size_t)g * MH + b] = 0.5f * (c63[b] + c64[b]); // q=0.5 interp
}

// ================= HGAT: masked double softmax + both einsums =================
__global__ __launch_bounds__(256) void k455_hgat_attn(const float* __restrict__ scores,
                                                      const float* __restrict__ sm,
                                                      const float* __restrict__ thr,
                                                      const float* __restrict__ vAll,
                                                      float* __restrict__ oF, _Float16* __restrict__ oH)
{
    int g = blockIdx.x, h = blockIdx.y, tid = threadIdx.x;
    __shared__ float sL[128][MH];
    __shared__ float vL[128][DHD];
    __shared__ float ef[MH][DHD];
    __shared__ float cmx[MH], csm[MH];
    for (int p = tid; p < 128 * MH; p += 256) {
        int b = p >> 5, m = p & 31;
        float sraw = scores[(((size_t)g * HH + h) * BB + b) * MH + m];
        bool inc = sm[((size_t)g * BB + b) * MH + m] >= thr[(size_t)g * MH + m];
        sL[b][m] = inc ? sraw : -1e9f;
    }
    for (int p = tid; p < 128 * DHD; p += 256) {
        int b = p >> 6, d = p & 63;
        vL[b][d] = vAll[((size_t)g * BB + b) * DD + h * DHD + d];
    }
    __syncthreads();
    if (tid < MH) {            // column (node-axis) softmax stats
        float mx = -1e30f;
        for (int b = 0; b < 128; ++b) mx = fmaxf(mx, sL[b][tid]);
        float s = 0.f;
        for (int b = 0; b < 128; ++b) s += expf(sL[b][tid] - mx);
        cmx[tid] = mx; csm[tid] = s;
    }
    __syncthreads();
    for (int p = tid; p < MH * DHD; p += 256) {  // efeat[m][d]
        int m = p >> 6, d = p & 63;
        float mx = cmx[m], acc = 0.f;
        for (int b = 0; b < 128; ++b) acc += expf(sL[b][m] - mx) * vL[b][d];
        ef[m][d] = acc / csm[m];
    }
    __syncthreads();
    int b = tid & 127, half = tid >> 7;          // 2 threads per node row
    float rmx = -1e30f;
    #pragma unroll
    for (int m = 0; m < MH; ++m) rmx = fmaxf(rmx, sL[b][m]);
    float w[MH]; float rs = 0.f;
    #pragma unroll
    for (int m = 0; m < MH; ++m) { float e = expf(sL[b][m] - rmx); w[m] = e; rs += e; }
    float inv = 1.0f / rs;
    for (int d = half * 32; d < half * 32 + 32; ++d) {
        float acc = 0.f;
        #pragma unroll
        for (int m = 0; m < MH; ++m) acc += w[m] * ef[m][d];
        acc *= inv;
        size_t idx = ((size_t)g * BB + b) * DD + h * DHD + d;
        oF[idx] = acc; oH[idx] = (_Float16)acc;
    }
}

// ================= head output de-normalize + permute =================
__global__ void k455_final(const float* __restrict__ dec, const float* __restrict__ means,
                           const float* __restrict__ stdev, float* __restrict__ out, int total)
{
    int idx = blockIdx.x * blockDim.x + threadIdx.x;
    if (idx >= total) return;
    int b = idx / (LL * CC);
    int rem = idx % (LL * CC);
    int lp = rem / CC, c = rem % CC;
    int n = lp / NOUT, o = lp % NOUT;
    int bc = b * CC + c;
    int t = bc * NP + n;
    out[idx] = dec[(size_t)t * NOUT + o] * stdev[bc] + means[bc];
}

// ================= host-side launch =================
static inline int cdiv(int a, int b) { return (a + b - 1) / b; }

static void launch_gemm(const _Float16* A, const _Float16* B, float* outF, _Float16* outH,
                        const float* bias, const float* res, int M, int N, int K, int gelu,
                        hipStream_t stream)
{
    dim3 g(cdiv(N, GBN), cdiv(M, GBM));
    k455_gemm_nt<<<g, dim3(256), 0, stream>>>(A, B, outF, outH, bias, res, M, N, K, gelu);
}

extern "C" void kernel_launch(void* const* d_in, const int* in_sizes, int n_in,
                              void* d_out, int out_size, void* d_ws, size_t ws_size,
                              hipStream_t stream)
{
    (void)in_sizes; (void)n_in; (void)out_size; (void)ws_size;
    const float* x_enc    = (const float*)d_in[0];
    const float* W_emb    = (const float*)d_in[4];
    const float* attn_w   = (const float*)d_in[5];
    const float* attn_b   = (const float*)d_in[6];
    const float* ln_g     = (const float*)d_in[7];
    const float* ln_b     = (const float*)d_in[8];
    const float* ffn_w1   = (const float*)d_in[9];
    const float* ffn_b1   = (const float*)d_in[10];
    const float* ffn_w2   = (const float*)d_in[11];
    const float* ffn_b2   = (const float*)d_in[12];
    const float* norm_g   = (const float*)d_in[13];
    const float* norm_b   = (const float*)d_in[14];
    const float* hedge    = (const float*)d_in[15];
    const float* hgat_wq  = (const float*)d_in[16];
    const float* hgat_wk  = (const float*)d_in[17];
    const float* hgat_wv  = (const float*)d_in[18];
    const float* hgat_wo  = (const float*)d_in[19];
    const float* hln_g    = (const float*)d_in[20];
    const float* hln_b    = (const float*)d_in[21];
    const float* head_w   = (const float*)d_in[22];
    const float* head_b   = (const float*)d_in[23];
    float* out = (float*)d_out;

    // ---- workspace layout ----
    char* p = (char*)d_ws;
    auto take = [&](size_t bytes) -> char* {
        char* r = p; p += (bytes + 255) & ~(size_t)255; return r;
    };
    float*     means   = (float*)take((size_t)BCN * 4);
    float*     stdev   = (float*)take((size_t)BCN * 4);
    _Float16*  XpH     = (_Float16*)take((size_t)TT * PP * 2);
    float*     bufX    = (float*)take((size_t)TT * DD * 4);
    _Float16*  bufXh   = (_Float16*)take((size_t)TT * DD * 2);
    float*     bufT    = (float*)take((size_t)TT * DD * 4);
    float*     bufQ    = (float*)take((size_t)TT * DD * 4);
    float*     bufK    = (float*)take((size_t)TT * DD * 4);
    float*     bufV    = (float*)take((size_t)TT * DD * 4);
    float*     bufO    = (float*)take((size_t)TT * DD * 4);
    _Float16*  bufOh   = (_Float16*)take((size_t)TT * DD * 2);
    _Float16*  y1H     = (_Float16*)take((size_t)TT * DFF_ * 2);
    float*     xall    = (float*)take((size_t)TT * DD * 4);
    _Float16*  xallH   = (_Float16*)take((size_t)TT * DD * 2);
    float*     scores  = (float*)take((size_t)(CC * NP) * HH * BB * MH * 4);
    float*     smbuf   = (float*)take((size_t)(CC * NP) * BB * MH * 4);
    float*     thrbuf  = (float*)take((size_t)(CC * NP) * MH * 4);
    float*     kEbuf   = (float*)take((size_t)MH * DD * 4);
    float*     decbuf  = (float*)take((size_t)TT * NOUT * 4);
    _Float16*  attnwH  = (_Float16*)take((size_t)ELN * 4 * DD * DD * 2);
    _Float16*  w1H     = (_Float16*)take((size_t)ELN * DFF_ * DD * 2);
    _Float16*  w2H     = (_Float16*)take((size_t)ELN * DD * DFF_ * 2);
    _Float16*  wembH   = (_Float16*)take((size_t)DD * PP * 2);
    _Float16*  hedgeH  = (_Float16*)take((size_t)MH * DD * 2);
    _Float16*  headwH  = (_Float16*)take((size_t)NOUT * DD * 2);
    _Float16*  wqT     = (_Float16*)take((size_t)DD * DD * 2);
    _Float16*  wkT     = (_Float16*)take((size_t)DD * DD * 2);
    _Float16*  wvT     = (_Float16*)take((size_t)DD * DD * 2);
    _Float16*  woT     = (_Float16*)take((size_t)DD * DD * 2);

    // ---- stats + patchify ----
    k455_stats<<<BB, 256, 0, stream>>>(x_enc, means, stdev);
    {
        int tot = TT * PP;
        k455_patchify<<<cdiv(tot, 256), 256, 0, stream>>>(x_enc, means, stdev, XpH, tot);
    }
    // ---- weight conversions ----
    {
        int n;
        n = ELN * 4 * DD * DD;  k455_cvt<<<cdiv(n, 256), 256, 0, stream>>>(attn_w, attnwH, n);
        n = ELN * DFF_ * DD;    k455_cvt<<<cdiv(n, 256), 256, 0, stream>>>(ffn_w1, w1H, n);
        n = ELN * DD * DFF_;    k455_cvt<<<cdiv(n, 256), 256, 0, stream>>>(ffn_w2, w2H, n);
        n = DD * PP;            k455_cvt<<<cdiv(n, 256), 256, 0, stream>>>(W_emb, wembH, n);
        n = MH * DD;            k455_cvt<<<cdiv(n, 256), 256, 0, stream>>>(hedge, hedgeH, n);
        n = NOUT * DD;          k455_cvt<<<cdiv(n, 256), 256, 0, stream>>>(head_w, headwH, n);
        n = DD * DD;
        k455_cvt_t<<<cdiv(n, 256), 256, 0, stream>>>(hgat_wq, wqT, DD, DD);
        k455_cvt_t<<<cdiv(n, 256), 256, 0, stream>>>(hgat_wk, wkT, DD, DD);
        k455_cvt_t<<<cdiv(n, 256), 256, 0, stream>>>(hgat_wv, wvT, DD, DD);
        k455_cvt_t<<<cdiv(n, 256), 256, 0, stream>>>(hgat_wo, woT, DD, DD);
    }
    // ---- embedding GEMM + pos emb ----
    launch_gemm(XpH, wembH, bufX, nullptr, nullptr, nullptr, TT, DD, PP, 0, stream);
    {
        int tot = TT * DD;
        k455_add_pos<<<cdiv(tot, 256), 256, 0, stream>>>(bufX, bufXh, tot);
    }
    // ---- decoder layers ----
    for (int l = 0; l < ELN; ++l) {
        const _Float16* wq = attnwH + (size_t)(l * 4 + 0) * DD * DD;
        const _Float16* wk = attnwH + (size_t)(l * 4 + 1) * DD * DD;
        const _Float16* wv = attnwH + (size_t)(l * 4 + 2) * DD * DD;
        const _Float16* wo = attnwH + (size_t)(l * 4 + 3) * DD * DD;
        const float* bq = attn_b + (size_t)(l * 4 + 0) * DD;
        const float* bk = attn_b + (size_t)(l * 4 + 1) * DD;
        const float* bv = attn_b + (size_t)(l * 4 + 2) * DD;
        const float* bo = attn_b + (size_t)(l * 4 + 3) * DD;
        launch_gemm(bufXh, wq, bufQ, nullptr, bq, nullptr, TT, DD, DD, 0, stream);
        launch_gemm(bufXh, wk, bufK, nullptr, bk, nullptr, TT, DD, DD, 0, stream);
        launch_gemm(bufXh, wv, bufV, nullptr, bv, nullptr, TT, DD, DD, 0, stream);
        k455_attn<<<dim3(BCN, HH), dim3(64), 0, stream>>>(bufQ, bufK, bufV, bufO, bufOh);
        launch_gemm(bufOh, wo, bufT, nullptr, bo, bufX, TT, DD, DD, 0, stream);  // x + o@Wo^T
        k455_ln<<<TT, 128, 0, stream>>>(bufT, ln_g + (size_t)(l * 2) * DD, ln_b + (size_t)(l * 2) * DD,
                                        bufX, bufXh, DD);
        launch_gemm(bufXh, w1H + (size_t)l * DFF_ * DD, nullptr, y1H,
                    ffn_b1 + (size_t)l * DFF_, nullptr, TT, DFF_, DD, 1, stream);  // gelu
        launch_gemm(y1H, w2H + (size_t)l * DD * DFF_, bufT, nullptr,
                    ffn_b2 + (size_t)l * DD, bufX, TT, DD, DFF_, 0, stream);       // x + ffn(x)
        k455_ln<<<TT, 128, 0, stream>>>(bufT, ln_g + (size_t)(l * 2 + 1) * DD,
                                        ln_b + (size_t)(l * 2 + 1) * DD, bufX, bufXh, DD);
    }
    // ---- final decoder LN -> tmp ----
    k455_ln<<<TT, 128, 0, stream>>>(bufX, norm_g, norm_b, bufT, nullptr, DD);
    // ---- HGAT ----
    {
        int tot = TT * DD;
        k455_gather<<<cdiv(tot, 256), 256, 0, stream>>>(bufT, xall, xallH, tot);
    }
    launch_gemm(xallH, wqT, bufQ, nullptr, nullptr, nullptr, TT, DD, DD, 0, stream);  // q = X@wq
    launch_gemm(xallH, wvT, bufV, nullptr, nullptr, nullptr, TT, DD, DD, 0, stream);  // v = X@wv
    launch_gemm(hedgeH, wkT, kEbuf, nullptr, nullptr, nullptr, MH, DD, DD, 0, stream); // k = E@wk
    k455_hgat_scores<<<dim3(CC * NP, HH), dim3(256), 0, stream>>>(bufQ, kEbuf, scores);
    k455_hgat_smthr<<<CC * NP, 128, 0, stream>>>(scores, smbuf, thrbuf);
    k455_hgat_attn<<<dim3(CC * NP, HH), dim3(256), 0, stream>>>(scores, smbuf, thrbuf, bufV, bufO, bufOh);
    launch_gemm(bufOh, woT, bufT, nullptr, nullptr, xall, TT, DD, DD, 0, stream);     // X + out@wo
    k455_ln<<<TT, 128, 0, stream>>>(bufT, hln_g, hln_b, bufX, nullptr, DD);           // Yall -> bufX
    {
        int tot = TT * DD;
        k455_scatter<<<cdiv(tot, 256), 256, 0, stream>>>(bufX, bufXh, tot);           // back to token order
    }
    // ---- head + de-normalize ----
    launch_gemm(bufXh, headwH, decbuf, nullptr, head_b, nullptr, TT, NOUT, DD, 0, stream);
    {
        int tot = BB * LL * CC;
        k455_final<<<cdiv(tot, 256), 256, 0, stream>>>(decbuf, means, stdev, out, tot);
    }
}
